// WaveletKANLayer_29996051595933
// MI455X (gfx1250) — matrix operand order repaired and checked by reading, not compiled, not run on hardware
//
#include <hip/hip_runtime.h>
#include <hip/hip_fp16.h>

typedef __attribute__((ext_vector_type(16))) _Float16 v16h;
typedef __attribute__((ext_vector_type(8)))  float    v8f;
typedef __attribute__((ext_vector_type(4)))  float    f4;

#define TPB 256

// ---------------------------------------------------------------------------
// Setup: slope[f] (Gumbel-softmax hard mix of Haar-pattern coefs),
// fused bias' = b_out + W_out @ b_nl, and B' = slope[k] * W_out[n,k]
// pre-packed in f16 WMMA B-fragment layout:
//   Bpack[((kt*NT + t)*32 + lane)*16 + p]  holds  B'[kt*32 + 8*(lane/16) + (p%8) + 16*(p/8),
//                                                    t*16 + (lane%16)]
// ---------------------------------------------------------------------------
__global__ void wkan_setup_kernel(const float* __restrict__ W_nl,
                                  const float* __restrict__ b_nl,
                                  const float* __restrict__ logits,
                                  const float* __restrict__ gumbel,
                                  const float* __restrict__ W_out,
                                  const float* __restrict__ b_out,
                                  float* __restrict__ slope_out,
                                  float* __restrict__ bias_out,
                                  _Float16* __restrict__ Bpack,
                                  int F, int O, int J)
{
    __shared__ float s_slope[1024]; // F <= 1024
    const int tid  = threadIdx.x;
    const int full = 1 << J;

    // --- per-feature slope ---
    for (int f = tid; f < F; f += TPB) {
        float coef[8];
        for (int j = 1; j <= J; ++j) {
            float c = W_nl[f * full + 0] * exp2f(-0.5f * (float)j); // cA_j
            int pos = 1;
            for (int k = 1; k <= j; ++k) {                          // cD_k blocks
                c += W_nl[f * full + pos] * exp2f(-0.5f * (float)k);
                pos += 1 << (j - k);
            }
            coef[j - 1] = c;
        }
        // gumbel softmax (tau=1, hard=True, straight-through)
        float y[8];
        float m = -3.402823466e38f;
        int am = 0;
        for (int j = 0; j < J; ++j) {
            y[j] = logits[j] + gumbel[f * J + j];
            if (y[j] > m) { m = y[j]; am = j; }   // first max, like jnp.argmax
        }
        float e[8], se = 0.f;
        for (int j = 0; j < J; ++j) { e[j] = expf(y[j] - m); se += e[j]; }
        float s = 0.f;
        for (int j = 0; j < J; ++j) {
            float soft = e[j] / se;
            float w = ((j == am) ? 1.f : 0.f) - soft + soft; // mimic ref arithmetic
            s += w * coef[j];
        }
        slope_out[f] = s;
        s_slope[f]   = s;
    }
    __syncthreads();

    // --- fused bias: b_out[n] + sum_k b_nl[k] * W_out[n,k] ---
    for (int n = tid; n < O; n += TPB) {
        float acc = b_out[n];
        for (int k = 0; k < F; ++k) acc += b_nl[k] * W_out[n * F + k];
        bias_out[n] = acc;
    }

    // --- pack B' into WMMA B-fragment layout (f16) ---
    const int NT = O >> 4;
    const int total = F * O;
    for (int i = tid; i < total; i += TPB) {
        int p    = i & 15;
        int lane = (i >> 4) & 31;
        int idx2 = i >> 9;
        int t    = idx2 % NT;
        int kt   = idx2 / NT;
        int K    = kt * 32 + ((lane >> 4) << 3) + (p & 7) + ((p >> 3) << 4);
        int Nn   = t * 16 + (lane & 15);
        Bpack[i] = (_Float16)(s_slope[K] * W_out[Nn * F + K]);
    }
}

// ---------------------------------------------------------------------------
// Specialized GEMM (F = KT*32 compile-time, O = 256): B' lives entirely in
// registers (KT*4 fragments = 256 VGPRs), grid-stride over 32-row M tiles.
// Inner loop = pure x streaming: 4 nontemporal b128 loads + 4 WMMAs per
// k-step, zero B traffic.
// ---------------------------------------------------------------------------
template<int KT>
__global__ __launch_bounds__(TPB)
void wkan_gemm_regB(const float* __restrict__ X,
                    const _Float16* __restrict__ Bpack,
                    const float* __restrict__ biasP,
                    float* __restrict__ Out,
                    int M, int O, int NT, int mtiles)
{
    const int lane = threadIdx.x & 31;
    const int wave = threadIdx.x >> 5;
    const int mw   = wave & 1;        // 2 M sub-tiles per block
    const int nw   = wave >> 1;       // 4 N groups of 64 columns
    const int h    = lane >> 4;
    const int lm   = lane & 15;
    const int F    = KT * 32;
    const int nt0  = nw * 4;

    // --- B' fragments resident in registers for the whole kernel ---
    v16h Bf[KT][4];
    #pragma unroll
    for (int kt = 0; kt < KT; ++kt)
        #pragma unroll
        for (int t = 0; t < 4; ++t)
            Bf[kt][t] = *(const v16h*)(Bpack +
                          ((size_t)(kt * NT + nt0 + t) * 32 + lane) * 16);

    float bias[4];
    #pragma unroll
    for (int t = 0; t < 4; ++t) bias[t] = biasP[(nt0 + t) * 16 + lm];

    for (int mt = blockIdx.x; mt < mtiles; mt += gridDim.x) {
        const int m0 = mt * 32 + mw * 16;
        int row = m0 + lm;
        if (row >= M) row = M - 1;            // clamp loads; stores guarded
        const float* __restrict__ xr = X + (size_t)row * F + 8 * h;

        v8f acc[4];
        acc[0] = v8f{0.f,0.f,0.f,0.f,0.f,0.f,0.f,0.f};
        acc[1] = acc[0]; acc[2] = acc[0]; acc[3] = acc[0];

        #pragma unroll
        for (int kt = 0; kt < KT; ++kt) {
            const float* p = xr + kt * 32;
            f4 a0 = __builtin_nontemporal_load((const f4*)(p));
            f4 a1 = __builtin_nontemporal_load((const f4*)(p + 4));
            f4 a2 = __builtin_nontemporal_load((const f4*)(p + 16));
            f4 a3 = __builtin_nontemporal_load((const f4*)(p + 20));
            v16h A;
            A[0]  = (_Float16)a0[0]; A[1]  = (_Float16)a0[1];
            A[2]  = (_Float16)a0[2]; A[3]  = (_Float16)a0[3];
            A[4]  = (_Float16)a1[0]; A[5]  = (_Float16)a1[1];
            A[6]  = (_Float16)a1[2]; A[7]  = (_Float16)a1[3];
            A[8]  = (_Float16)a2[0]; A[9]  = (_Float16)a2[1];
            A[10] = (_Float16)a2[2]; A[11] = (_Float16)a2[3];
            A[12] = (_Float16)a3[0]; A[13] = (_Float16)a3[1];
            A[14] = (_Float16)a3[2]; A[15] = (_Float16)a3[3];

            acc[0] = __builtin_amdgcn_wmma_f32_16x16x32_f16(false, A, false,
                        Bf[kt][0], (short)0, acc[0], false, false);
            acc[1] = __builtin_amdgcn_wmma_f32_16x16x32_f16(false, A, false,
                        Bf[kt][1], (short)0, acc[1], false, false);
            acc[2] = __builtin_amdgcn_wmma_f32_16x16x32_f16(false, A, false,
                        Bf[kt][2], (short)0, acc[2], false, false);
            acc[3] = __builtin_amdgcn_wmma_f32_16x16x32_f16(false, A, false,
                        Bf[kt][3], (short)0, acc[3], false, false);
        }

        // Epilogue: C/D layout -> VGPR r: M = m0 + 8h + r, N = lm + 16*tile
        const int mbase = m0 + 8 * h;
        #pragma unroll
        for (int t = 0; t < 4; ++t) {
            const int nn = (nt0 + t) * 16 + lm;
            #pragma unroll
            for (int r = 0; r < 8; ++r) {
                const int mm = mbase + r;
                if (mm < M)
                    __builtin_nontemporal_store(acc[t][r] + bias[t],
                                                &Out[(size_t)mm * O + nn]);
            }
        }
    }
}

// ---------------------------------------------------------------------------
// Generic fallback GEMM (any F%32==0, O == 256): B fragments loaded per k-step.
// ---------------------------------------------------------------------------
__global__ __launch_bounds__(TPB)
void wkan_gemm_generic(const float* __restrict__ X,
                       const _Float16* __restrict__ Bpack,
                       const float* __restrict__ biasP,
                       float* __restrict__ Out,
                       int M, int F, int O)
{
    const int lane = threadIdx.x & 31;
    const int wave = threadIdx.x >> 5;
    const int mw   = wave & 1;
    const int nw   = wave >> 1;
    const int NT   = O >> 4;
    const int KT   = F >> 5;
    const int h    = lane >> 4;
    const int lm   = lane & 15;

    const int m0 = blockIdx.x * 32 + mw * 16;
    int row = m0 + lm;
    if (row >= M) row = M - 1;
    const float* __restrict__ xrow = X + (size_t)row * F;
    const int nt0 = nw * 4;

    v8f acc[4];
    acc[0] = v8f{0.f,0.f,0.f,0.f,0.f,0.f,0.f,0.f};
    acc[1] = acc[0]; acc[2] = acc[0]; acc[3] = acc[0];

    for (int kt = 0; kt < KT; ++kt) {
        const int kb = kt * 32 + 8 * h;
        f4 a0 = *(const f4*)(xrow + kb);
        f4 a1 = *(const f4*)(xrow + kb + 4);
        f4 a2 = *(const f4*)(xrow + kb + 16);
        f4 a3 = *(const f4*)(xrow + kb + 20);
        v16h A;
        A[0]  = (_Float16)a0[0]; A[1]  = (_Float16)a0[1];
        A[2]  = (_Float16)a0[2]; A[3]  = (_Float16)a0[3];
        A[4]  = (_Float16)a1[0]; A[5]  = (_Float16)a1[1];
        A[6]  = (_Float16)a1[2]; A[7]  = (_Float16)a1[3];
        A[8]  = (_Float16)a2[0]; A[9]  = (_Float16)a2[1];
        A[10] = (_Float16)a2[2]; A[11] = (_Float16)a2[3];
        A[12] = (_Float16)a3[0]; A[13] = (_Float16)a3[1];
        A[14] = (_Float16)a3[2]; A[15] = (_Float16)a3[3];

        const _Float16* __restrict__ bb =
            Bpack + ((size_t)(kt * NT + nt0) * 32 + lane) * 16;
        v16h B0 = *(const v16h*)(bb);
        v16h B1 = *(const v16h*)(bb + 32 * 16);
        v16h B2 = *(const v16h*)(bb + 2 * 32 * 16);
        v16h B3 = *(const v16h*)(bb + 3 * 32 * 16);

        acc[0] = __builtin_amdgcn_wmma_f32_16x16x32_f16(false, A, false, B0,
                                                        (short)0, acc[0], false, false);
        acc[1] = __builtin_amdgcn_wmma_f32_16x16x32_f16(false, A, false, B1,
                                                        (short)0, acc[1], false, false);
        acc[2] = __builtin_amdgcn_wmma_f32_16x16x32_f16(false, A, false, B2,
                                                        (short)0, acc[2], false, false);
        acc[3] = __builtin_amdgcn_wmma_f32_16x16x32_f16(false, A, false, B3,
                                                        (short)0, acc[3], false, false);
    }

    const int mbase = m0 + 8 * h;
    #pragma unroll
    for (int t = 0; t < 4; ++t) {
        const int nn = (nt0 + t) * 16 + lm;
        const float bias = biasP[nn];
        #pragma unroll
        for (int r = 0; r < 8; ++r) {
            const int mm = mbase + r;
            if (mm < M) Out[(size_t)mm * O + nn] = acc[t][r] + bias;
        }
    }
}

// ---------------------------------------------------------------------------
extern "C" void kernel_launch(void* const* d_in, const int* in_sizes, int n_in,
                              void* d_out, int out_size, void* d_ws, size_t ws_size,
                              hipStream_t stream) {
    const float* x      = (const float*)d_in[0];
    const float* W_nl   = (const float*)d_in[1];
    const float* b_nl   = (const float*)d_in[2];
    const float* logits = (const float*)d_in[3];
    const float* gumbel = (const float*)d_in[4];
    const float* W_out  = (const float*)d_in[5];
    const float* b_out  = (const float*)d_in[6];

    const int J = in_sizes[3];           // 3
    const int F = in_sizes[2];           // 256
    const int O = in_sizes[6];           // 256
    const int M = in_sizes[0] / F;       // B*N = 131072

    float*    wsf   = (float*)d_ws;
    float*    slope = wsf;                            // F floats
    float*    biasP = wsf + F;                        // O floats
    _Float16* Bpack = (_Float16*)(wsf + F + O);       // F*O halfs (128 KB)

    wkan_setup_kernel<<<1, TPB, 0, stream>>>(W_nl, b_nl, logits, gumbel,
                                             W_out, b_out, slope, biasP, Bpack,
                                             F, O, J);

    if (F == 256 && O == 256) {
        const int mtiles = (M + 31) / 32;
        int grid = mtiles < 1024 ? mtiles : 1024;   // ~4 tiles/block amortization
        wkan_gemm_regB<8><<<grid, TPB, 0, stream>>>(x, Bpack, biasP,
                                                    (float*)d_out, M, O,
                                                    O >> 4, mtiles);
    } else {
        const int grid = (M + 31) / 32;
        wkan_gemm_generic<<<grid, TPB, 0, stream>>>(x, Bpack, biasP,
                                                    (float*)d_out, M, F, O);
    }
}
